// NonLocal4Point_69063074120024
// MI455X (gfx1250) — matrix operand order repaired and checked by reading, not compiled, not run on hardware
//
#include <hip/hip_runtime.h>

// ---------------------------------------------------------------------------
// NonLocal attention block on gfx1250 (MI455X).
// bf16 WMMA everywhere, flash-style streaming softmax, workgroup-cooperative
// async global->LDS staging of K/V tiles (double buffered), fused out-proj.
// ---------------------------------------------------------------------------

typedef __bf16 bf16;
typedef __attribute__((ext_vector_type(8)))  float v8f;
typedef __attribute__((ext_vector_type(16))) bf16  v16bf;
typedef __attribute__((ext_vector_type(8)))  bf16  v8bf;

#define DEVI __device__ __forceinline__

DEVI v8f vzero8() {
    v8f v;
#pragma unroll
    for (int i = 0; i < 8; ++i) v[i] = 0.0f;
    return v;
}

// D = A(16x32 bf16) * B(32x16 bf16) + C(16x16 f32)
DEVI v8f wmma_bf16(v16bf a, v16bf b, v8f c) {
    return __builtin_amdgcn_wmma_f32_16x16x32_bf16(
        false, a, false, b, (short)0, c, false, false);
}

// A-fragment (16x32 bf16) from row-major matrix (stride ld elements).
// lane: row = base+lane%16, h = lane/16; elems 0..15 = K(16h..16h+15)
// contiguous  -> two contiguous 16B loads.
DEVI v16bf ldA(const bf16* base, int row, int ld, int k0) {
    int lane = threadIdx.x & 31;
    const bf16* p = base + (size_t)(row + (lane & 15)) * ld + k0 + 16 * (lane >> 4);
    v8bf lo = *(const v8bf*)p;
    v8bf hi = *(const v8bf*)(p + 8);
    v16bf a;
#pragma unroll
    for (int i = 0; i < 8; ++i) { a[i] = lo[i]; a[8 + i] = hi[i]; }
    return a;
}

// B-fragment (32x16 bf16) from K-contiguous (Ncols x K) row-major storage.
// lane: col = base+lane%16; elems 0..15 = K(16h..16h+15) contiguous (32B).
DEVI v16bf ldB(const bf16* base, int col, int ld, int k0) {
    int lane = threadIdx.x & 31;
    const bf16* p = base + (size_t)(col + (lane & 15)) * ld + k0 + 16 * (lane >> 4);
    v8bf lo = *(const v8bf*)p;
    v8bf hi = *(const v8bf*)(p + 8);
    v16bf b;
#pragma unroll
    for (int i = 0; i < 8; ++i) { b[i] = lo[i]; b[8 + i] = hi[i]; }
    return b;
}

// Build a B-fragment (32 K-rows x 16 cols) from two f32 C-tiles holding the
// same 32x16 block (c0 rows 0..15, c1 rows 16..31). Half-wave swap only.
DEVI v16bf packB(v8f c0, v8f c1) {
    int lane = threadIdx.x & 31;
    bool hi = lane >= 16;
    v16bf b;
#pragma unroll
    for (int r = 0; r < 8; ++r) {
        float sw0 = __shfl_xor(c0[r], 16, 32);
        float sw1 = __shfl_xor(c1[r], 16, 32);
        float lo  = hi ? sw1 : c0[r];   // K = 16h + r
        float hh  = hi ? c1[r] : sw0;   // K = 16h + 8 + r
        b[r]     = (bf16)lo;
        b[8 + r] = (bf16)hh;
    }
    return b;
}

// async 16B global -> LDS copy (one lane = one 16B chunk), ASYNCcnt tracked
DEVI void async_cp16(unsigned lds_addr, unsigned long long gaddr) {
    asm volatile("global_load_async_to_lds_b128 %0, %1, off"
                 :: "v"(lds_addr), "v"(gaddr) : "memory");
}
DEVI void wait_async0() {
    asm volatile("s_wait_asynccnt 0x0" ::: "memory");
}

// ---------------------------------------------------------------------------
// prep kernels
// ---------------------------------------------------------------------------
__global__ void __launch_bounds__(256) cvt_bf16_kernel(
    const float* __restrict__ in, bf16* __restrict__ out, int n) {
    int i = blockIdx.x * blockDim.x + threadIdx.x;
    if (i < n) out[i] = (bf16)in[i];
}

__global__ void __launch_bounds__(256) bn_prep_kernel(
    const float* __restrict__ bw, const float* __restrict__ gamma,
    const float* __restrict__ beta, const float* __restrict__ mean,
    const float* __restrict__ var, float* __restrict__ fscale,
    float* __restrict__ fshift, int C) {
    int i = blockIdx.x * blockDim.x + threadIdx.x;
    if (i < C) {
        float inv = rsqrtf(var[i] + 1e-5f);
        float fs  = gamma[i] * inv;
        fscale[i] = fs;
        fshift[i] = fs * (bw[i] - mean[i]) + beta[i];
    }
}

// transpose + convert: X (B,C,N) f32 -> XT (B,N,C) bf16; 32x32 LDS tiles
__global__ void __launch_bounds__(256) transpose_cvt_kernel(
    const float* __restrict__ X, bf16* __restrict__ XT, int C, int N) {
    __shared__ bf16 tile[32][33];
    int b  = blockIdx.z;
    int c0 = blockIdx.y * 32, n0 = blockIdx.x * 32;
#pragma unroll
    for (int i = threadIdx.y; i < 32; i += 8)
        tile[i][threadIdx.x] =
            (bf16)X[((size_t)b * C + c0 + i) * N + n0 + threadIdx.x];
    __syncthreads();
#pragma unroll
    for (int i = threadIdx.y; i < 32; i += 8)
        XT[((size_t)b * N + n0 + i) * C + c0 + threadIdx.x] = tile[threadIdx.x][i];
}

// ---------------------------------------------------------------------------
// proj_nt: OUT (B,N,Ci) bf16 = XT (B,N,C) @ W^T + bias     (Q and K^T)
// all operands bf16, fully contiguous fragment loads; one wave per 16x16 tile
// ---------------------------------------------------------------------------
__global__ void __launch_bounds__(256) proj_nt(
    const bf16* __restrict__ XT, const bf16* __restrict__ W,   // W: (Ci,C)
    const float* __restrict__ bias, bf16* __restrict__ out,
    int B, int C, int N, int Ci) {
    int wid  = (blockIdx.x * blockDim.x + threadIdx.x) >> 5;
    int lane = threadIdx.x & 31;
    int nci = Ci / 16, nnt = N / 16;
    int cit = wid % nci;
    int nt  = (wid / nci) % nnt;
    int b   = wid / (nci * nnt);
    int h   = lane >> 4;
    int ci  = cit * 16 + (lane & 15);

    const bf16* xb = XT + (size_t)b * N * C;
    v8f acc = vzero8();
    for (int kb = 0; kb < C; kb += 32) {
        v16bf a  = ldA(xb, nt * 16, C, kb);   // rows n, K=c contiguous
        v16bf bf = ldB(W, cit * 16, C, kb);   // cols ci, K=c contiguous
        acc = wmma_bf16(a, bf, acc);
    }
    float bv = bias[ci];
#pragma unroll
    for (int r = 0; r < 8; ++r) {
        int row = nt * 16 + r + 8 * h;
        out[((size_t)b * N + row) * Ci + ci] = (bf16)(acc[r] + bv);
    }
}

// ---------------------------------------------------------------------------
// proj_tn: OUT (B,Ci,N) bf16 = W (Ci,C) @ X (C,N) + bias   (V^T layout)
// A from W rows, B from XT rows (both contiguous bf16)
// ---------------------------------------------------------------------------
__global__ void __launch_bounds__(256) proj_tn(
    const bf16* __restrict__ XT, const bf16* __restrict__ W,
    const float* __restrict__ bias, bf16* __restrict__ out,
    int B, int C, int N, int Ci) {
    int wid  = (blockIdx.x * blockDim.x + threadIdx.x) >> 5;
    int lane = threadIdx.x & 31;
    int nci = Ci / 16, nnt = N / 16;
    int nt  = wid % nnt;
    int cit = (wid / nnt) % nci;
    int b   = wid / (nci * nnt);
    int h   = lane >> 4;
    int n   = nt * 16 + (lane & 15);

    const bf16* xb = XT + (size_t)b * N * C;
    v8f acc = vzero8();
    for (int kb = 0; kb < C; kb += 32) {
        v16bf a  = ldA(W, cit * 16, C, kb);    // rows ci, K=c contiguous
        v16bf bf = ldB(xb, nt * 16, C, kb);    // cols n,  K=c contiguous
        acc = wmma_bf16(a, bf, acc);
    }
#pragma unroll
    for (int r = 0; r < 8; ++r) {
        int ci = cit * 16 + r + 8 * h;
        out[((size_t)b * Ci + ci) * N + n] = (bf16)(acc[r] + bias[ci]);
    }
}

// ---------------------------------------------------------------------------
// Fused flash attention + output projection.
// Block = 8 waves sharing K/V M-tiles via async-to-LDS double buffering.
// Wave w owns query rows [16*(grp*8+w) .. +16).
//   S^T tile = K_tile(32xCi) @ Q^T        (16 wmma / tile, A from LDS)
//   online softmax: per-lane stats + one shfl_xor(16)
//   y^T (Ci x 16) += V^T_tile @ P^T       (16 wmma / tile, A from LDS)
//   out = Ww @ y, folded BN + ReLU        (8 wmma / 16-channel chunk)
// ---------------------------------------------------------------------------
#define KP 264   // 256 + 8 pad (bank rotate)
#define VP 40    // 32 + 8 pad

__global__ void __launch_bounds__(256) attn_kernel(
    const bf16* __restrict__ Q,    // (B,N,Ci)
    const bf16* __restrict__ KT,   // (B,M,Ci)
    const bf16* __restrict__ VT,   // (B,Ci,M)
    const bf16* __restrict__ WwBf, // (Co,Ci)
    const float* __restrict__ fscale, const float* __restrict__ fshift,
    float* __restrict__ out,       // (B,Co,N) f32
    int B, int Co, int Ci, int N) {
    __shared__ bf16 Ksh[2][32][KP];
    __shared__ bf16 Vsh[2][256][VP];

    int lane = threadIdx.x & 31;
    int wave = threadIdx.x >> 5;
    int grp  = blockIdx.x & 31;          // 32 q-groups per batch
    int b    = blockIdx.x >> 5;
    int qb   = grp * 8 + wave;
    int h    = lane >> 4;

    const bf16* Qb = Q  + (size_t)b * N * Ci;
    const bf16* Kb = KT + (size_t)b * N * Ci;
    const bf16* Vb = VT + (size_t)b * Ci * N;

    // cooperative async stage of one (K,V) M-tile into LDS buffer `buf`
    auto stage = [&](int buf, int mt) {
        int tid = threadIdx.x;
#pragma unroll
        for (int i = tid; i < 1024; i += 256) {        // K: 32 rows x 512B
            int r = i >> 5, o = (i & 31) * 8;
            async_cp16((unsigned)(size_t)&Ksh[buf][r][o],
                       (unsigned long long)(size_t)(Kb + (size_t)(mt + r) * Ci + o));
        }
#pragma unroll
        for (int i = tid; i < 1024; i += 256) {        // V: 256 rows x 64B
            int r = i >> 2, o = (i & 3) * 8;
            async_cp16((unsigned)(size_t)&Vsh[buf][r][o],
                       (unsigned long long)(size_t)(Vb + (size_t)r * N + mt + o));
        }
    };

    // Q block as 8 B-fragments, held in registers for the whole pass
    v16bf qf[8];
#pragma unroll
    for (int kk = 0; kk < 8; ++kk) qf[kk] = ldB(Qb, qb * 16, Ci, kk * 32);

    v8f yac[16];                       // y^T accumulators: 16 ci-tiles x 16 q
#pragma unroll
    for (int t = 0; t < 16; ++t) yac[t] = vzero8();
    float mrun = -1e30f, lsum = 0.0f;

    stage(0, 0);
    for (int mt = 0; mt < N; mt += 32) {
        int buf = (mt >> 5) & 1;
        wait_async0();                 // own async writes done
        __syncthreads();               // => everyone's writes done
        if (mt + 32 < N) stage(buf ^ 1, mt + 32);

        const bf16* kbase = &Ksh[buf][0][0];
        const bf16* vbase = &Vsh[buf][0][0];

        v8f s0 = vzero8(), s1 = vzero8();
#pragma unroll
        for (int kk = 0; kk < 8; ++kk) {
            v16bf a0 = ldA(kbase, 0, KP, kk * 32);
            s0 = wmma_bf16(a0, qf[kk], s0);
            v16bf a1 = ldA(kbase, 16, KP, kk * 32);
            s1 = wmma_bf16(a1, qf[kk], s1);
        }
        // online softmax; lane pair (L, L^16) shares query row q = L%16
        float tmax = -1e30f;
#pragma unroll
        for (int r = 0; r < 8; ++r) {
            tmax = fmaxf(tmax, s0[r]);
            tmax = fmaxf(tmax, s1[r]);
        }
        tmax = fmaxf(tmax, __shfl_xor(tmax, 16, 32));
        float mnew  = fmaxf(mrun, tmax);
        float alpha = __expf(mrun - mnew);
        lsum *= alpha;
#pragma unroll
        for (int t = 0; t < 16; ++t) yac[t] *= alpha;
        v8f p0, p1;
#pragma unroll
        for (int r = 0; r < 8; ++r) {
            p0[r] = __expf(s0[r] - mnew);
            p1[r] = __expf(s1[r] - mnew);
            lsum += p0[r] + p1[r];
        }
        mrun = mnew;
        v16bf pb = packB(p0, p1);      // P^T as B-fragment (K = 32 m-rows)
#pragma unroll
        for (int t = 0; t < 16; ++t) {
            v16bf av = ldA(vbase, t * 16, VP, 0);
            yac[t] = wmma_bf16(av, pb, yac[t]);
        }
        __syncthreads();
    }

    float lfull = lsum + __shfl_xor(lsum, 16, 32);
    float rinv  = 1.0f / lfull;
#pragma unroll
    for (int t = 0; t < 16; ++t) yac[t] *= rinv;

    // y^T -> 8 B-fragments over Ci (frees accumulator VGPRs)
    v16bf yb[8];
#pragma unroll
    for (int kk = 0; kk < 8; ++kk) yb[kk] = packB(yac[2 * kk], yac[2 * kk + 1]);

    // fused output projection + BN + ReLU
    int n = qb * 16 + (lane & 15);
    for (int cb = 0; cb < Co; cb += 16) {
        v8f acc = vzero8();
#pragma unroll
        for (int kk = 0; kk < 8; ++kk) {
            v16bf aw = ldA(WwBf, cb, Ci, kk * 32);
            acc = wmma_bf16(aw, yb[kk], acc);
        }
#pragma unroll
        for (int r = 0; r < 8; ++r) {
            int co  = cb + r + 8 * h;
            float v = acc[r] * fscale[co] + fshift[co];
            out[((size_t)b * Co + co) * N + n] = fmaxf(v, 0.0f);
        }
    }
}

// ---------------------------------------------------------------------------
extern "C" void kernel_launch(void* const* d_in, const int* in_sizes, int n_in,
                              void* d_out, int out_size, void* d_ws, size_t ws_size,
                              hipStream_t stream) {
    const float* qfeat = (const float*)d_in[0];
    const float* kfeat = (const float*)d_in[1];
    const float* Wq = (const float*)d_in[2];
    const float* bq = (const float*)d_in[3];
    const float* Wk = (const float*)d_in[4];
    const float* bk = (const float*)d_in[5];
    const float* Wg = (const float*)d_in[6];
    const float* bg = (const float*)d_in[7];
    const float* Ww = (const float*)d_in[8];
    const float* bw = (const float*)d_in[9];
    const float* gamma = (const float*)d_in[10];
    const float* beta  = (const float*)d_in[11];
    const float* mean  = (const float*)d_in[12];
    const float* var   = (const float*)d_in[13];
    float* out = (float*)d_out;

    const int B = 4, C = 512, Ci = 256, N = 4096;
    const size_t xtBytes   = (size_t)B * N * C * 2;    // 16.78 MB each
    const size_t projBytes = (size_t)B * N * Ci * 2;   // 8.39 MB each
    const size_t wBytes    = (size_t)Ci * C * 2;       // 0.26 MB each

    char* ws = (char*)d_ws;
    bf16* XTq = (bf16*)(ws);                    ws += xtBytes;
    bf16* XTk = (bf16*)(ws);                    ws += xtBytes;
    bf16* Qb  = (bf16*)(ws);                    ws += projBytes;
    bf16* KTb = (bf16*)(ws);                    ws += projBytes;
    bf16* VTb = (bf16*)(ws);                    ws += projBytes;
    bf16* Wqb = (bf16*)(ws);                    ws += wBytes;
    bf16* Wkb = (bf16*)(ws);                    ws += wBytes;
    bf16* Wgb = (bf16*)(ws);                    ws += wBytes;
    bf16* Wwb = (bf16*)(ws);                    ws += wBytes;
    float* fscale = (float*)(ws);               ws += (size_t)C * 4;
    float* fshift = (float*)(ws);

    // weight conversion + BN folding
    int wn = Ci * C;
    cvt_bf16_kernel<<<(wn + 255) / 256, 256, 0, stream>>>(Wq, Wqb, wn);
    cvt_bf16_kernel<<<(wn + 255) / 256, 256, 0, stream>>>(Wk, Wkb, wn);
    cvt_bf16_kernel<<<(wn + 255) / 256, 256, 0, stream>>>(Wg, Wgb, wn);
    cvt_bf16_kernel<<<(wn + 255) / 256, 256, 0, stream>>>(Ww, Wwb, wn);
    bn_prep_kernel<<<(C + 255) / 256, 256, 0, stream>>>(
        bw, gamma, beta, mean, var, fscale, fshift, C);

    // transpose-convert inputs: (B,C,N) f32 -> (B,N,C) bf16
    dim3 tb(32, 8), tg(N / 32, C / 32, B);
    transpose_cvt_kernel<<<tg, tb, 0, stream>>>(qfeat, XTq, C, N);
    transpose_cvt_kernel<<<tg, tb, 0, stream>>>(kfeat, XTk, C, N);

    // projections (bf16 WMMA, contiguous fragments)
    int projWaves = B * (N / 16) * (Ci / 16);          // 16384 waves
    proj_nt<<<projWaves / 8, 256, 0, stream>>>(XTq, Wqb, bq, Qb,  B, C, N, Ci);
    proj_nt<<<projWaves / 8, 256, 0, stream>>>(XTk, Wkb, bk, KTb, B, C, N, Ci);
    proj_tn<<<projWaves / 8, 256, 0, stream>>>(XTk, Wgb, bg, VTb, B, C, N, Ci);

    // fused attention + output projection: 128 blocks x 8 waves
    attn_kernel<<<B * 32, 256, 0, stream>>>(
        Qb, KTb, VTb, Wwb, fscale, fshift, out, B, C, Ci, N);
}
